// SparseExpertModule_61761629716683
// MI455X (gfx1250) — matrix operand order repaired and checked by reading, not compiled, not run on hardware
//
#include <hip/hip_runtime.h>

typedef __attribute__((ext_vector_type(16))) __bf16 v16bf;
typedef __attribute__((ext_vector_type(8)))  float  v8f;

#define TOK  16384
#define DDIM 1024
#define HDIM 2048
#define NEXP 4
#define MROW 32          // token rows per workgroup (2 x 16-row WMMA tiles)
#define KT1  (DDIM/32)   // 32 k-tiles for stage 1
#define NT1  (HDIM/16)   // 128 n-tiles for stage 1
#define KT2  (HDIM/32)   // 64 k-tiles for stage 2
#define NT2  (DDIM/16)   // 64 n-tiles for stage 2

union Frag { uint4 u[2]; v16bf v; };
union V8B  { uint4 u; __bf16 b[8]; };

__device__ __forceinline__ float wave_sum(float v) {
#pragma unroll
  for (int m = 16; m >= 1; m >>= 1) v += __shfl_xor(v, m, 32);
  return v;
}

__device__ __forceinline__ v8f wmma_bf16(v16bf a, v16bf b, v8f c) {
  return __builtin_amdgcn_wmma_f32_16x16x32_bf16(false, a, false, b, (short)0, c,
                                                 false, false);
}

// ---------------------------------------------------------------- fp32 -> bf16
__global__ void k_cvt_bf16(const float* __restrict__ src, __bf16* __restrict__ dst, int n) {
  int i = blockIdx.x * 256 + threadIdx.x;
  if (i < n) dst[i] = (__bf16)src[i];
}

// ------------------------------------------------- repack w1 [E,D,H] -> tiles
// dst element idx = ((e*KT1+kt)*NT1+nt)*512 + L*16 + j
// holds  w1[e][kt*32 + (L>=16?8:0) + (j&7) + (j>=8?16:0)][nt*16 + (L&15)]   (B-matrix lane layout)
__global__ void k_repack_w1(const float* __restrict__ w1, __bf16* __restrict__ w1p) {
  int idx  = blockIdx.x * 256 + threadIdx.x;
  int j    = idx & 15;
  int L    = (idx >> 4) & 31;
  int tile = idx >> 9;
  int nt   = tile % NT1;
  int rest = tile / NT1;
  int kt   = rest % KT1;
  int e    = rest / KT1;
  int k    = kt * 32 + (L >> 4) * 8 + (j & 7) + (j >> 3) * 16;
  int n    = nt * 16 + (L & 15);
  w1p[idx] = (__bf16)w1[((size_t)e * DDIM + k) * HDIM + n];
}

// ------------------------------------------------- repack w2 [E,H,D] -> tiles
__global__ void k_repack_w2(const float* __restrict__ w2, __bf16* __restrict__ w2p) {
  int idx  = blockIdx.x * 256 + threadIdx.x;
  int j    = idx & 15;
  int L    = (idx >> 4) & 31;
  int tile = idx >> 9;
  int nt   = tile % NT2;
  int rest = tile / NT2;
  int kt   = rest % KT2;
  int e    = rest / KT2;
  int k    = kt * 32 + (L >> 4) * 8 + (j & 7) + (j >> 3) * 16;
  int n    = nt * 16 + (L & 15);
  w2p[idx] = (__bf16)w2[((size_t)e * HDIM + k) * DDIM + n];
}

// --------------------------------------------------------- router: 1 wave/token
__global__ __launch_bounds__(256) void k_router(
    const float* __restrict__ h, const float* __restrict__ rw,
    const float* __restrict__ rng, const float* __restrict__ rnb,
    float* __restrict__ gates) {
  const int lane  = threadIdx.x & 31;
  const int wv    = threadIdx.x >> 5;
  const int token = blockIdx.x * 8 + wv;

  const float* hr = h + (size_t)token * DDIM;
  float hv[32];
#pragma unroll
  for (int i = 0; i < 32; ++i) hv[i] = hr[i * 32 + lane];

  float s = 0.f, sq = 0.f;
#pragma unroll
  for (int i = 0; i < 32; ++i) { s += hv[i]; sq += hv[i] * hv[i]; }
  s = wave_sum(s); sq = wave_sum(sq);
  float mu = s * (1.f / DDIM);
  float rs = rsqrtf(sq * (1.f / DDIM) - mu * mu + 1e-5f);

#pragma unroll
  for (int i = 0; i < 32; ++i) {
    int c = i * 32 + lane;
    hv[i] = (hv[i] - mu) * rs * rng[c] + rnb[c];
  }

  float lg[4];
#pragma unroll
  for (int e = 0; e < 4; ++e) {
    float p = 0.f;
#pragma unroll
    for (int i = 0; i < 32; ++i) p += hv[i] * rw[(size_t)e * DDIM + i * 32 + lane];
    lg[e] = wave_sum(p);
  }

  float mx = fmaxf(fmaxf(lg[0], lg[1]), fmaxf(lg[2], lg[3]));
  float gv[4], ssum = 0.f;
#pragma unroll
  for (int e = 0; e < 4; ++e) { gv[e] = expf(lg[e] - mx); ssum += gv[e]; }
#pragma unroll
  for (int e = 0; e < 4; ++e) gv[e] /= ssum;

  float m0 = gv[0]; int a = 0;
  if (gv[1] > m0) { m0 = gv[1]; a = 1; }
  if (gv[2] > m0) { m0 = gv[2]; a = 2; }
  if (gv[3] > m0) { m0 = gv[3]; a = 3; }
  float m1 = -1.f; int bsel = -1;
#pragma unroll
  for (int e = 0; e < 4; ++e) {
    bool c = (e != a) && (gv[e] > m1);
    m1 = c ? gv[e] : m1;
    bsel = c ? e : bsel;
  }
  float den = m0 + m1 + 1e-8f;
  if (lane < 4) {
    float val = (lane == a) ? m0 / den : ((lane == bsel) ? m1 / den : 0.f);
    gates[(size_t)token * NEXP + lane] = val;
  }
}

// ----------------------------------------------------- fused MoE FFN + LNs
// One workgroup = 32 token rows (two 16-row WMMA A-tiles); each B fragment
// feeds two WMMAs. Inner loops are pointer-walked so all per-k loads use
// constant immediate offsets (1 pointer add per stream per k-step).
__global__ __launch_bounds__(256) void k_moe(
    const __bf16* __restrict__ hb,  const __bf16* __restrict__ w1p,
    const __bf16* __restrict__ w2p, const float* __restrict__ gates,
    const float* __restrict__ ln_g, const float* __restrict__ ln_b,
    const float* __restrict__ on_g, const float* __restrict__ on_b,
    float* __restrict__ out) {
  extern __shared__ char smem[];
  __bf16* hidb  = (__bf16*)smem;                               // 32*2048 bf16 (128K)
  float*  outac = (float*)(smem + MROW * HDIM * 2);            // 32*1024 f32  (128K)
  float*  smean = (float*)(smem + MROW * HDIM * 2 + MROW * DDIM * 4);
  float*  srstd = smean + MROW;
  float*  gat   = srstd + MROW;

  const int tid  = threadIdx.x;
  const int wv   = tid >> 5;
  const int lane = tid & 31;
  const int lidx = lane & 15;
  const int hsel = lane >> 4;
  const int koff = hsel * 8;
  const int r0   = blockIdx.x * MROW;

  for (int i = tid * 4; i < MROW * DDIM; i += 1024)
    *(float4*)(outac + i) = float4{0.f, 0.f, 0.f, 0.f};

  const __bf16* arow0 = hb + (size_t)(r0 + lidx) * DDIM + koff;
  const __bf16* arow1 = hb + (size_t)(r0 + 16 + lidx) * DDIM + koff;

  for (int e = 0; e < NEXP; ++e) {
    // ---- stage 1: hid = h_blk(32xD) * w1[e](DxH), bf16 WMMA, f32 acc ----
#pragma unroll 1
    for (int g = 0; g < 4; ++g) {
      v8f acc[2][4] = {};
      const __bf16* pa0 = arow0;
      const __bf16* pa1 = arow1;
      const __bf16* pb  = w1p +
          ((size_t)e * KT1 * NT1 + (size_t)(wv * 16 + g * 4)) * 512 + lane * 16;
      for (int k = 0; k < KT1; ++k) {
        Frag a0, a1;
        a0.u[0] = *(const uint4*)(pa0);
        a0.u[1] = *(const uint4*)(pa0 + 16);
        a1.u[0] = *(const uint4*)(pa1);
        a1.u[1] = *(const uint4*)(pa1 + 16);
#pragma unroll
        for (int t = 0; t < 4; ++t) {
          v16bf b = *(const v16bf*)(pb + (size_t)t * 512);
          acc[0][t] = wmma_bf16(a0.v, b, acc[0][t]);
          acc[1][t] = wmma_bf16(a1.v, b, acc[1][t]);
        }
        pa0 += 32;
        pa1 += 32;
        pb  += (size_t)NT1 * 512;
      }
#pragma unroll
      for (int m = 0; m < 2; ++m)
#pragma unroll
        for (int t = 0; t < 4; ++t) {
          int n0 = (wv * 16 + g * 4 + t) * 16 + lidx;
#pragma unroll
          for (int r = 0; r < 8; ++r)
            hidb[(m * 16 + r + hsel * 8) * HDIM + n0] = (__bf16)acc[m][t][r];
        }
    }
    if (tid < MROW) gat[tid] = gates[(size_t)(r0 + tid) * NEXP + e];
    __syncthreads();
    // ---- per-row LN stats over H (vectorized 8 bf16 / load) ----
#pragma unroll
    for (int sub = 0; sub < 4; ++sub) {
      int row = wv * 4 + sub;
      float s = 0.f, sq = 0.f;
      for (int i = lane * 8; i < HDIM; i += 32 * 8) {
        V8B v; v.u = *(const uint4*)(hidb + row * HDIM + i);
#pragma unroll
        for (int q = 0; q < 8; ++q) { float x = (float)v.b[q]; s += x; sq += x * x; }
      }
      s = wave_sum(s); sq = wave_sum(sq);
      if (lane == 0) {
        float mu = s * (1.f / HDIM);
        smean[row] = mu;
        srstd[row] = rsqrtf(sq * (1.f / HDIM) - mu * mu + 1e-5f);
      }
    }
    __syncthreads();
    // ---- normalize + relu, back to bf16 in LDS (8-wide) ----
    for (int i0 = tid * 8; i0 < MROW * HDIM; i0 += 256 * 8) {
      int row = i0 >> 11, col = i0 & (HDIM - 1);
      V8B v; v.u = *(const uint4*)(hidb + i0);
      float4 g0 = *(const float4*)(ln_g + (size_t)e * HDIM + col);
      float4 g1 = *(const float4*)(ln_g + (size_t)e * HDIM + col + 4);
      float4 b0 = *(const float4*)(ln_b + (size_t)e * HDIM + col);
      float4 b1 = *(const float4*)(ln_b + (size_t)e * HDIM + col + 4);
      float mu = smean[row], rstd = srstd[row];
      float gg[8] = {g0.x, g0.y, g0.z, g0.w, g1.x, g1.y, g1.z, g1.w};
      float bb[8] = {b0.x, b0.y, b0.z, b0.w, b1.x, b1.y, b1.z, b1.w};
#pragma unroll
      for (int q = 0; q < 8; ++q) {
        float y = ((float)v.b[q] - mu) * rstd * gg[q] + bb[q];
        v.b[q] = (__bf16)fmaxf(y, 0.f);
      }
      *(uint4*)(hidb + i0) = v.u;
    }
    __syncthreads();
    // ---- stage 2: out_e = hid(32xH) * w2[e](HxD), gated accumulate ----
#pragma unroll 1
    for (int g = 0; g < 2; ++g) {
      v8f acc[2][4] = {};
      const __bf16* pa0 = hidb + lidx * HDIM + koff;
      const __bf16* pa1 = hidb + (16 + lidx) * HDIM + koff;
      const __bf16* pb  = w2p +
          ((size_t)e * KT2 * NT2 + (size_t)(wv * 8 + g * 4)) * 512 + lane * 16;
      for (int k = 0; k < KT2; ++k) {
        Frag a0, a1;
        a0.u[0] = *(const uint4*)(pa0);
        a0.u[1] = *(const uint4*)(pa0 + 16);
        a1.u[0] = *(const uint4*)(pa1);
        a1.u[1] = *(const uint4*)(pa1 + 16);
#pragma unroll
        for (int t = 0; t < 4; ++t) {
          v16bf b = *(const v16bf*)(pb + (size_t)t * 512);
          acc[0][t] = wmma_bf16(a0.v, b, acc[0][t]);
          acc[1][t] = wmma_bf16(a1.v, b, acc[1][t]);
        }
        pa0 += 32;
        pa1 += 32;
        pb  += (size_t)NT2 * 512;
      }
#pragma unroll
      for (int m = 0; m < 2; ++m)
#pragma unroll
        for (int t = 0; t < 4; ++t) {
          int n0 = (wv * 8 + g * 4 + t) * 16 + lidx;
#pragma unroll
          for (int r = 0; r < 8; ++r) {
            int row = m * 16 + r + hsel * 8;
            outac[row * DDIM + n0] += gat[row] * acc[m][t][r];
          }
        }
    }
    __syncthreads();
  }
  // ---- final LN over D (vectorized), write fp32 output ----
#pragma unroll
  for (int sub = 0; sub < 4; ++sub) {
    int row = wv * 4 + sub;
    float s = 0.f, sq = 0.f;
    for (int i = lane * 4; i < DDIM; i += 32 * 4) {
      float4 x = *(const float4*)(outac + row * DDIM + i);
      s += x.x + x.y + x.z + x.w;
      sq += x.x * x.x + x.y * x.y + x.z * x.z + x.w * x.w;
    }
    s = wave_sum(s); sq = wave_sum(sq);
    if (lane == 0) {
      float mu = s * (1.f / DDIM);
      smean[row] = mu;
      srstd[row] = rsqrtf(sq * (1.f / DDIM) - mu * mu + 1e-5f);
    }
  }
  __syncthreads();
  for (int i0 = tid * 4; i0 < MROW * DDIM; i0 += 1024) {
    int row = i0 >> 10, col = i0 & (DDIM - 1);
    float4 x = *(const float4*)(outac + i0);
    float4 g = *(const float4*)(on_g + col);
    float4 b = *(const float4*)(on_b + col);
    float mu = smean[row], rstd = srstd[row];
    float4 r;
    r.x = (x.x - mu) * rstd * g.x + b.x;
    r.y = (x.y - mu) * rstd * g.y + b.y;
    r.z = (x.z - mu) * rstd * g.z + b.z;
    r.w = (x.w - mu) * rstd * g.w + b.w;
    *(float4*)(out + (size_t)(r0 + row) * DDIM + col) = r;
  }
}

extern "C" void kernel_launch(void* const* d_in, const int* in_sizes, int n_in,
                              void* d_out, int out_size, void* d_ws, size_t ws_size,
                              hipStream_t stream) {
  const float* h   = (const float*)d_in[0];
  const float* rw  = (const float*)d_in[1];
  const float* rng = (const float*)d_in[2];
  const float* rnb = (const float*)d_in[3];
  const float* w1  = (const float*)d_in[4];
  const float* lng = (const float*)d_in[5];
  const float* lnb = (const float*)d_in[6];
  const float* w2  = (const float*)d_in[7];
  const float* ong = (const float*)d_in[8];
  const float* onb = (const float*)d_in[9];
  float* out = (float*)d_out;

  char* ws = (char*)d_ws;
  __bf16* hb    = (__bf16*)ws;                          // 32 MB
  __bf16* w1p   = (__bf16*)(ws + (size_t)33554432);     // 16 MB
  __bf16* w2p   = (__bf16*)(ws + (size_t)50331648);     // 16 MB
  float*  gates = (float*)(ws + (size_t)67108864);      // 256 KB

  k_cvt_bf16<<<(TOK * DDIM) / 256, 256, 0, stream>>>(h, hb, TOK * DDIM);
  k_repack_w1<<<(NEXP * DDIM * HDIM) / 256, 256, 0, stream>>>(w1, w1p);
  k_repack_w2<<<(NEXP * DDIM * HDIM) / 256, 256, 0, stream>>>(w2, w2p);
  k_router<<<TOK / 8, 256, 0, stream>>>(h, rw, rng, rnb, gates);

  size_t smem = (size_t)MROW * HDIM * 2 + (size_t)MROW * DDIM * 4 + 3 * MROW * 4;
  k_moe<<<TOK / MROW, 256, smem, stream>>>(hb, w1p, w2p, gates,
                                           lng, lnb, ong, onb, out);
}
